// CuGraphDistributedGINE_30520037606039
// MI455X (gfx1250) — matrix operand order repaired and checked by reading, not compiled, not run on hardware
//
#include <hip/hip_runtime.h>
#include <hip/hip_bf16.h>
#include <stdint.h>

typedef __bf16  v16bf __attribute__((ext_vector_type(16)));
typedef float   v8f   __attribute__((ext_vector_type(8)));

constexpr int kN  = 100000;   // nodes
constexpr int kE  = 1600000;  // edges
constexpr int kD  = 128;      // channels
constexpr int kED = 16;       // edge_dim
constexpr int kL  = 3;        // layers
constexpr int kG  = 1024;     // graphs
constexpr int kC  = 10;       // classes

// ---------- helpers ----------
static __device__ __forceinline__ unsigned short f2bf(float f) {
    union { float f; unsigned u; } c; c.f = f;
    unsigned u = c.u;
    unsigned r = u + 0x7FFFu + ((u >> 16) & 1u);   // RNE
    return (unsigned short)(r >> 16);
}
static __device__ __forceinline__ unsigned pack2(float a, float b) {
    return (unsigned)f2bf(a) | ((unsigned)f2bf(b) << 16);
}
static __device__ __forceinline__ v8f wmma_bf16(v16bf a, v16bf b, v8f c) {
    // D = A(16x32 bf16) * B(32x16 bf16) + C(16x16 f32)
    return __builtin_amdgcn_wmma_f32_16x16x32_bf16(false, a, false, b, (short)0, c, false, false);
}
// Load a pre-packed B fragment (v16bf) for (kchunk, ntile, lane): 8 dwords contiguous.
static __device__ __forceinline__ v16bf loadB(const unsigned* __restrict__ Wp, int kc, int nt, int lane) {
    const unsigned* p = Wp + (unsigned)((((kc * 8 + nt) * 32) + lane) << 3);
    union { v16bf v; uint4 q[2]; } r;
    r.q[0] = *(const uint4*)p;
    r.q[1] = *(const uint4*)(p + 4);
    return r.v;
}

// ---------- weight packing: f32 [K x 128] row-major -> bf16 B-fragment layout ----------
// dst[((kc*8 + ntile)*32 + lane)*8 + j] = pack2(W[k][n], W[k+1][n])
// n = ntile*16 + lane%16 ; k = kc*32 + (lane/16)*8 + 2*j + (j>=4 ? 8 : 0) ; rows >= Ksrc are zero (K padding)
__global__ void pack_w_kernel(const float* __restrict__ src, unsigned* __restrict__ dst,
                              int Ksrc, int Kchunks) {
    int t = blockIdx.x * blockDim.x + threadIdx.x;
    int total = Kchunks * 8 * 32 * 8;
    if (t >= total) return;
    int j  = t & 7;
    int l  = (t >> 3) & 31;
    int nt = (t >> 8) & 7;
    int kc = t >> 11;
    int n = nt * 16 + (l & 15);
    int k = kc * 32 + (l >> 4) * 8 + 2 * j + (j & 4) * 2;
    float v0 = (k     < Ksrc) ? src[(unsigned)(k       * kD + n)] : 0.f;
    float v1 = (k + 1 < Ksrc) ? src[(unsigned)((k + 1) * kD + n)] : 0.f;
    dst[t] = pack2(v0, v1);
}

// ---------- edge kernel: e = eattr @ We + be ; m = relu(h[src]+e) ; agg[dst] += m ----------
// One wave handles 16 edges x 128 channels; WMMA for the 16x(32pad)x128 GEMM.
// All gather/scatter offsets are u32 (tables are 51.2 MB) -> SGPR-base + 32-bit voffset.
__global__ __launch_bounds__(256) void edge_kernel(
    const float* __restrict__ h, const float* __restrict__ eattr,
    const unsigned* __restrict__ Wep, const float* __restrict__ be,
    const int* __restrict__ srcI, const int* __restrict__ dstI,
    float* __restrict__ agg)
{
    int lane = threadIdx.x & 31;
    int wave = threadIdx.x >> 5;
    long e0 = ((long)blockIdx.x * 8 + wave) * 16;
    if (e0 >= kE) return;

    int row  = lane & 15;   // A-row / C-column-within-tile for this lane
    int half = lane >> 4;

    // A fragment: lanes 0-15 hold K 0..7 (+16..23 zero), lanes 16-31 hold K 8..15 (+24..31 zero)
    union { v16bf v; unsigned u[8]; } A;
    const float4* ar = (const float4*)(eattr + (e0 + row) * kED + half * 8);
    float4 a0 = ar[0], a1 = ar[1];
    A.u[0] = pack2(a0.x, a0.y); A.u[1] = pack2(a0.z, a0.w);
    A.u[2] = pack2(a1.x, a1.y); A.u[3] = pack2(a1.z, a1.w);
    A.u[4] = A.u[5] = A.u[6] = A.u[7] = 0u;

    // this lane's output rows are M = j + 8*half ; load the 8 indices as 2x int4
    const int4* sp = (const int4*)(srcI + e0 + 8 * half);
    const int4* dp = (const int4*)(dstI + e0 + 8 * half);
    int4 s0 = sp[0], s1 = sp[1];
    int4 d0 = dp[0], d1 = dp[1];
    unsigned sb[8] = { (unsigned)s0.x << 7, (unsigned)s0.y << 7, (unsigned)s0.z << 7, (unsigned)s0.w << 7,
                       (unsigned)s1.x << 7, (unsigned)s1.y << 7, (unsigned)s1.z << 7, (unsigned)s1.w << 7 };
    unsigned db[8] = { (unsigned)d0.x << 7, (unsigned)d0.y << 7, (unsigned)d0.z << 7, (unsigned)d0.w << 7,
                       (unsigned)d1.x << 7, (unsigned)d1.y << 7, (unsigned)d1.z << 7, (unsigned)d1.w << 7 };

    #pragma unroll
    for (int t = 0; t < 8; ++t) {
        unsigned n = (unsigned)(t * 16 + row);
        float bias = be[n];
        v8f acc;
        #pragma unroll
        for (int j = 0; j < 8; ++j) acc[j] = bias;
        acc = wmma_bf16(A.v, loadB(Wep, 0, t, lane), acc);
        #pragma unroll
        for (int j = 0; j < 8; ++j) {
            float v = acc[j] + h[sb[j] + n];
            v = v > 0.f ? v : 0.f;
            atomicAdd(&agg[db[j] + n], v);
        }
    }
}

// ---------- node kernel: out = relu( BNReLU((h+agg)@W1+b1) @ W2 + b2 ) ----------
// One wave = 16 nodes; per-wave 4KB LDS tile stages the A operands (bf16) and the
// C-layout -> A-layout transpose between the two GEMMs.
__global__ __launch_bounds__(256) void node_kernel(
    const float* __restrict__ h, float* __restrict__ agg /* in: aggr, out: new h */,
    const unsigned* __restrict__ W1p, const float* __restrict__ b1,
    const float* __restrict__ gamma, const float* __restrict__ beta,
    const unsigned* __restrict__ W2p, const float* __restrict__ b2)
{
    __shared__ __align__(16) unsigned lds[8][1024];   // 8 waves x (16x128 bf16)
    int lane = threadIdx.x & 31;
    int wave = threadIdx.x >> 5;
    int tile_i = blockIdx.x * 8 + wave;
    bool active = tile_i < (kN / 16);
    if (!active) tile_i = 0;                          // clamped waves keep EXEC uniform, no stores
    unsigned base = (unsigned)tile_i * 16u * 128u;    // element offset of this 16-row tile
    unsigned* tile = lds[wave];

    // phase 1: tile = bf16(h + agg), each lane packs half a row (64 floats) via float4
    {
        int r  = lane >> 1;
        int cb = (lane & 1) * 64;
        unsigned ro = base + (unsigned)(r * kD + cb);
        const float4* hp = (const float4*)(h   + ro);
        const float4* ap = (const float4*)(agg + ro);
        unsigned* tp = tile + (r * kD + cb) / 2;
        #pragma unroll
        for (int q = 0; q < 16; ++q) {
            float4 hv = hp[q], av = ap[q];
            tp[2 * q]     = pack2(hv.x + av.x, hv.y + av.y);
            tp[2 * q + 1] = pack2(hv.z + av.z, hv.w + av.w);
        }
    }
    __syncthreads();

    int row = lane & 15, half = lane >> 4;
    union { v16bf v; uint4 q[2]; } a[4];
    #pragma unroll
    for (int c = 0; c < 4; ++c) {                      // A-layout reads (ds_load_b128 x2)
        const unsigned* bp = tile + row * 64 + c * 16 + half * 4;
        a[c].q[0] = *(const uint4*)bp;
        a[c].q[1] = *(const uint4*)(bp + 8);
    }
    __syncthreads();

    const float rstd = rsqrtf(1.0f + 1e-5f);
    unsigned short* t16 = (unsigned short*)tile;

    // phase 2: GEMM1 + BN + ReLU -> bf16 back into LDS (transposes C-layout to memory order)
    #pragma unroll
    for (int t = 0; t < 8; ++t) {
        int n = t * 16 + row;
        float bias = b1[n];
        v8f acc;
        #pragma unroll
        for (int j = 0; j < 8; ++j) acc[j] = bias;
        #pragma unroll
        for (int c = 0; c < 4; ++c) acc = wmma_bf16(a[c].v, loadB(W1p, c, t, lane), acc);
        float sc = rstd * gamma[n], sh = beta[n];
        #pragma unroll
        for (int j = 0; j < 8; ++j) {
            float v = acc[j] * sc + sh;
            v = v > 0.f ? v : 0.f;
            t16[(j + 8 * half) * kD + n] = f2bf(v);
        }
    }
    __syncthreads();

    // phase 3: reload A from intermediate, GEMM2 + ReLU -> global (overwrites agg rows we own)
    #pragma unroll
    for (int c = 0; c < 4; ++c) {
        const unsigned* bp = tile + row * 64 + c * 16 + half * 4;
        a[c].q[0] = *(const uint4*)bp;
        a[c].q[1] = *(const uint4*)(bp + 8);
    }
    #pragma unroll
    for (int t = 0; t < 8; ++t) {
        int n = t * 16 + row;
        float bias = b2[n];
        v8f acc;
        #pragma unroll
        for (int j = 0; j < 8; ++j) acc[j] = bias;
        #pragma unroll
        for (int c = 0; c < 4; ++c) acc = wmma_bf16(a[c].v, loadB(W2p, c, t, lane), acc);
        if (active) {
            unsigned ob = base + (unsigned)(half * 8 * kD + n);   // j*kD folds into imm offset
            #pragma unroll
            for (int j = 0; j < 8; ++j) {
                float v = acc[j];
                agg[ob + (unsigned)(j * kD)] = v > 0.f ? v : 0.f;
            }
        }
    }
}

// ---------- pooling: pooled[batch[v]] += h[v] ----------
__global__ void pool_kernel(const float* __restrict__ h, const int* __restrict__ batch,
                            float* __restrict__ pooled)
{
    long t = (long)blockIdx.x * blockDim.x + threadIdx.x;
    if (t >= (long)kN * kD) return;
    unsigned node = (unsigned)(t >> 7);
    unsigned ch   = (unsigned)(t & 127);
    atomicAdd(&pooled[((unsigned)batch[node] << 7) + ch], h[(unsigned)t]);
}

// ---------- classifier GEMM1: Y = relu(X @ W + b), X [nrows x 128] ----------
__global__ __launch_bounds__(256) void rowgemm_relu_kernel(
    const float* __restrict__ X, const unsigned* __restrict__ Wp,
    const float* __restrict__ bias, float* __restrict__ Y)
{
    __shared__ __align__(16) unsigned lds[8][1024];
    int lane = threadIdx.x & 31;
    int wave = threadIdx.x >> 5;
    unsigned base = (unsigned)(blockIdx.x * 8 + wave) * 16u * 128u;
    unsigned* tile = lds[wave];
    {
        int r = lane >> 1, cb = (lane & 1) * 64;
        const float4* xp = (const float4*)(X + base + (unsigned)(r * kD + cb));
        unsigned* tp = tile + (r * kD + cb) / 2;
        #pragma unroll
        for (int q = 0; q < 16; ++q) {
            float4 xv = xp[q];
            tp[2 * q]     = pack2(xv.x, xv.y);
            tp[2 * q + 1] = pack2(xv.z, xv.w);
        }
    }
    __syncthreads();
    int row = lane & 15, half = lane >> 4;
    union { v16bf v; uint4 q[2]; } a[4];
    #pragma unroll
    for (int c = 0; c < 4; ++c) {
        const unsigned* bp = tile + row * 64 + c * 16 + half * 4;
        a[c].q[0] = *(const uint4*)bp;
        a[c].q[1] = *(const uint4*)(bp + 8);
    }
    #pragma unroll
    for (int t = 0; t < 8; ++t) {
        int n = t * 16 + row;
        float b = bias[n];
        v8f acc;
        #pragma unroll
        for (int j = 0; j < 8; ++j) acc[j] = b;
        #pragma unroll
        for (int c = 0; c < 4; ++c) acc = wmma_bf16(a[c].v, loadB(Wp, c, t, lane), acc);
        unsigned ob = base + (unsigned)(half * 8 * kD + n);
        #pragma unroll
        for (int j = 0; j < 8; ++j) {
            float v = acc[j];
            Y[ob + (unsigned)(j * kD)] = v > 0.f ? v : 0.f;
        }
    }
}

// ---------- classifier GEMM2: out = X @ Wc2 + bc2   (128 -> 10, tiny, VALU) ----------
__global__ void cls2_kernel(const float* __restrict__ X, const float* __restrict__ Wc2,
                            const float* __restrict__ bc2, float* __restrict__ out)
{
    int t = blockIdx.x * blockDim.x + threadIdx.x;
    if (t >= kG * kC) return;
    int g = t / kC, c = t % kC;
    float s = bc2[c];
    const float* x = X + (unsigned)(g * kD);
    #pragma unroll 4
    for (int k = 0; k < kD; ++k) s += x[k] * Wc2[(unsigned)(k * kC + c)];
    out[t] = s;
}

// ---------- launch ----------
extern "C" void kernel_launch(void* const* d_in, const int* in_sizes, int n_in,
                              void* d_out, int out_size, void* d_ws, size_t ws_size,
                              hipStream_t stream)
{
    (void)in_sizes; (void)n_in; (void)out_size; (void)ws_size;
    const float* x     = (const float*)d_in[0];
    const float* eattr = (const float*)d_in[1];
    const float* We    = (const float*)d_in[2];
    const float* be    = (const float*)d_in[3];
    const float* W1    = (const float*)d_in[4];
    const float* b1    = (const float*)d_in[5];
    const float* gamma = (const float*)d_in[6];
    const float* beta  = (const float*)d_in[7];
    const float* W2    = (const float*)d_in[8];
    const float* b2    = (const float*)d_in[9];
    const float* Wc1   = (const float*)d_in[10];
    const float* bc1   = (const float*)d_in[11];
    const float* Wc2   = (const float*)d_in[12];
    const float* bc2   = (const float*)d_in[13];
    const int*   eidx  = (const int*)d_in[14];
    const int*   batch = (const int*)d_in[15];
    const int* srcI = eidx;
    const int* dstI = eidx + kE;

    size_t off = 0;
    auto take = [&](size_t bytes) -> void* {
        void* p = (char*)d_ws + off;
        off += (bytes + 255) & ~(size_t)255;
        return p;
    };
    float* bufA   = (float*)take((size_t)kN * kD * 4);
    float* bufB   = (float*)take((size_t)kN * kD * 4);
    float* pooled = (float*)take((size_t)kG * kD * 4);
    float* cls    = (float*)take((size_t)kG * kD * 4);
    unsigned* Wep[kL]; unsigned* W1p[kL]; unsigned* W2p[kL];
    for (int i = 0; i < kL; ++i) Wep[i] = (unsigned*)take(2048 * 4);   // K padded to 32
    for (int i = 0; i < kL; ++i) W1p[i] = (unsigned*)take(8192 * 4);
    for (int i = 0; i < kL; ++i) W2p[i] = (unsigned*)take(8192 * 4);
    unsigned* Wc1p = (unsigned*)take(8192 * 4);

    // pack weights into WMMA B-fragment layout (bf16)
    for (int i = 0; i < kL; ++i) {
        pack_w_kernel<<<8,  256, 0, stream>>>(We + (size_t)i * kED * kD, Wep[i], kED, 1);
        pack_w_kernel<<<32, 256, 0, stream>>>(W1 + (size_t)i * kD * kD,  W1p[i], kD, 4);
        pack_w_kernel<<<32, 256, 0, stream>>>(W2 + (size_t)i * kD * kD,  W2p[i], kD, 4);
    }
    pack_w_kernel<<<32, 256, 0, stream>>>(Wc1, Wc1p, kD, 4);

    // three GINE layers, ping-ponging bufA/bufB
    const float* h = x;
    float* aggOf[kL] = { bufA, bufB, bufA };
    for (int i = 0; i < kL; ++i) {
        float* agg = aggOf[i];
        hipMemsetAsync(agg, 0, (size_t)kN * kD * 4, stream);
        edge_kernel<<<kE / 128, 256, 0, stream>>>(h, eattr, Wep[i], be + i * kD,
                                                  srcI, dstI, agg);
        node_kernel<<<(kN / 16 + 7) / 8, 256, 0, stream>>>(h, agg, W1p[i], b1 + i * kD,
                                                           gamma + i * kD, beta + i * kD,
                                                           W2p[i], b2 + i * kD);
        h = agg;
    }

    // global_add_pool + classifier
    hipMemsetAsync(pooled, 0, (size_t)kG * kD * 4, stream);
    pool_kernel<<<(int)(((size_t)kN * kD + 255) / 256), 256, 0, stream>>>(h, batch, pooled);
    rowgemm_relu_kernel<<<kG / 128, 256, 0, stream>>>(pooled, Wc1p, bc1, cls);
    cls2_kernel<<<(kG * kC + 255) / 256, 256, 0, stream>>>(cls, Wc2, bc2, (float*)d_out);
}
